// HierarchicalFormationTransformer_58772332478797
// MI455X (gfx1250) — matrix operand order repaired and checked by reading, not compiled, not run on hardware
//
#include <hip/hip_runtime.h>
#include <hip/hip_bf16.h>
#include <math.h>
#include <stdint.h>

typedef __bf16 bf16;
typedef bf16 v16bf __attribute__((ext_vector_type(16)));
typedef bf16 v8bf  __attribute__((ext_vector_type(8)));
typedef float v8f  __attribute__((ext_vector_type(8)));

#define Lc 6
#define Bc 4
#define Sc 1024
#define Ec 1024
#define Hc 16
#define Fc 4096
#define DHc 64

enum { EPI_BF16 = 0, EPI_BF16_GELU = 1, EPI_F32_RESID = 2 };

// ---------------------------------------------------------------------------
// CDNA5 async global->LDS copy (tracked by ASYNCcnt, bypasses VGPRs).
// ISA 15.18.3: GLOBAL_LOAD_ASYNC_TO_LDS_B128, GV mode (VDST = per-lane LDS
// address, VADDR = 64-bit global address, SADDR = off/NULL).
// ---------------------------------------------------------------------------
__device__ __forceinline__ void async_ld_b128(void* lds, const void* g) {
  unsigned l = (unsigned)(uintptr_t)lds;  // low 32 bits of generic ptr = LDS addr
  asm volatile("global_load_async_to_lds_b128 %0, %1, off"
               :: "v"(l), "v"(g) : "memory");
}
__device__ __forceinline__ void wait_async0() {
  asm volatile("s_wait_asynccnt 0x0" ::: "memory");
}

// ---------------------------------------------------------------------------
// bias_sum = spatial + directional   (head-invariant, computed once)
// ---------------------------------------------------------------------------
__global__ __launch_bounds__(256) void add_bias_k(const float* __restrict__ a,
                                                  const float* __restrict__ b,
                                                  float* __restrict__ o, int n) {
  int i = blockIdx.x * blockDim.x + threadIdx.x;
  if (i < n) o[i] = a[i] + b[i];
}

// ---------------------------------------------------------------------------
// Transpose + convert: WT_bf16[N][K] = (bf16) W_f32[K][N].
// ---------------------------------------------------------------------------
__global__ __launch_bounds__(256) void transpose_cvt_k(const float* __restrict__ W,
                                                       bf16* __restrict__ WT,
                                                       int K, int N) {
  __shared__ float sT[64 * 65];
  const int k0 = blockIdx.x * 64, n0 = blockIdx.y * 64;
  const int tid = threadIdx.x;
#pragma unroll
  for (int i = 0; i < 4; ++i) {
    int c = tid + i * 256;
    int kr = c >> 4, n4 = (c & 15) << 2;
    float4 w = *(const float4*)(W + (size_t)(k0 + kr) * N + n0 + n4);
    sT[kr * 65 + n4 + 0] = w.x;
    sT[kr * 65 + n4 + 1] = w.y;
    sT[kr * 65 + n4 + 2] = w.z;
    sT[kr * 65 + n4 + 3] = w.w;
  }
  __syncthreads();
#pragma unroll
  for (int i = 0; i < 2; ++i) {
    int c = tid + i * 256;
    int nr = c >> 3, k8 = (c & 7) << 3;
    v8bf o;
#pragma unroll
    for (int j = 0; j < 8; ++j) o[j] = (bf16)sT[(k8 + j) * 65 + nr];
    *(v8bf*)(WT + (size_t)(n0 + nr) * K + k0 + k8) = o;
  }
}

// ---------------------------------------------------------------------------
// LayerNorm over E=1024; one block per row.
// ---------------------------------------------------------------------------
template <int OUT_BF16>
__global__ __launch_bounds__(256) void layernorm_k(const float* __restrict__ x,
                                                   const float* __restrict__ g,
                                                   const float* __restrict__ b,
                                                   void* __restrict__ outp) {
  const int row = blockIdx.x;
  const int tid = threadIdx.x;
  const int lane = tid & 31, wid = tid >> 5;
  float4 xv = *(const float4*)(x + (size_t)row * Ec + tid * 4);
  float s  = xv.x + xv.y + xv.z + xv.w;
  float sq = xv.x * xv.x + xv.y * xv.y + xv.z * xv.z + xv.w * xv.w;
#pragma unroll
  for (int m = 16; m >= 1; m >>= 1) {
    s += __shfl_xor(s, m, 32);
    sq += __shfl_xor(sq, m, 32);
  }
  __shared__ float red[16];
  if (lane == 0) { red[wid] = s; red[8 + wid] = sq; }
  __syncthreads();
  float S = 0.f, Q = 0.f;
#pragma unroll
  for (int w = 0; w < 8; ++w) { S += red[w]; Q += red[8 + w]; }
  float mean = S * (1.0f / Ec);
  float var  = Q * (1.0f / Ec) - mean * mean;
  float rstd = rsqrtf(var + 1e-5f);
  float xs[4] = {xv.x, xv.y, xv.z, xv.w};
#pragma unroll
  for (int j = 0; j < 4; ++j) {
    int e = tid * 4 + j;
    float y = (xs[j] - mean) * rstd * g[e] + b[e];
    if (OUT_BF16) ((bf16*)outp)[(size_t)row * Ec + e] = (bf16)y;
    else          ((float*)outp)[(size_t)row * Ec + e] = y;
  }
}

// ---------------------------------------------------------------------------
// GEMM: out[M,N] = A_bf16[M,K] @ WT_bf16[N,K]^T + bias (+epilogue)
// 128x128x32 tile, 8 waves (2x4), wave tile 64x32 = 8 WMMA accums.
// Double-buffered via CDNA5 async global->LDS loads: next k-block streams into
// the other buffer while WMMAs run; only s_wait_asynccnt 0 + barrier at the end.
// ---------------------------------------------------------------------------
template <int EPI>
__global__ __launch_bounds__(256) void gemm_bf16(
    const bf16* __restrict__ A, const bf16* __restrict__ WT,
    const float* __restrict__ bias, const float* __restrict__ resid,
    void* __restrict__ outp, int M, int N, int K) {
  __shared__ bf16 sA[2][128 * 40];   // [m][k], stride 40 keeps 16B alignment
  __shared__ bf16 sB[2][128 * 40];   // [n][k]
  const int tid = threadIdx.x;
  const int lane = tid & 31, wid = tid >> 5;
  const int wm = wid >> 2, wn = wid & 3;     // 2x4 wave grid
  const int bm = blockIdx.y * 128, bn = blockIdx.x * 128;

  v8f acc[4][2] = {};

  const bf16* gA = A + (size_t)bm * K;
  const bf16* gB = WT + (size_t)bn * K;
  const int nk = K >> 5;

  // prologue: stage k-block 0 into buffer 0 (async)
#pragma unroll
  for (int i = 0; i < 2; ++i) {
    int c = tid + i * 256;
    int row = c >> 2, c8 = (c & 3) << 3;
    async_ld_b128(&sA[0][row * 40 + c8], gA + (size_t)row * K + c8);
    async_ld_b128(&sB[0][row * 40 + c8], gB + (size_t)row * K + c8);
  }
  wait_async0();
  __syncthreads();

  for (int kt = 0; kt < nk; ++kt) {
    const int cur = kt & 1, nb = cur ^ 1;
    if (kt + 1 < nk) {                      // stream next k-block (async)
      int k0 = (kt + 1) << 5;
#pragma unroll
      for (int i = 0; i < 2; ++i) {
        int c = tid + i * 256;
        int row = c >> 2, c8 = (c & 3) << 3;
        async_ld_b128(&sA[nb][row * 40 + c8], gA + (size_t)row * K + k0 + c8);
        async_ld_b128(&sB[nb][row * 40 + c8], gB + (size_t)row * K + k0 + c8);
      }
    }
    // compute on buffer `cur` while async loads fly
    v16bf af[4];
#pragma unroll
    for (int mt = 0; mt < 4; ++mt) {
      int row = wm * 64 + mt * 16 + (lane & 15);
      int base = (lane & 16) ? 8 : 0;
      v8bf lo = *(const v8bf*)(&sA[cur][row * 40 + base]);
      v8bf hi = *(const v8bf*)(&sA[cur][row * 40 + base + 16]);
#pragma unroll
      for (int j = 0; j < 8; ++j) { af[mt][j] = lo[j]; af[mt][j + 8] = hi[j]; }
    }
#pragma unroll
    for (int nt = 0; nt < 2; ++nt) {
      int n = wn * 32 + nt * 16 + (lane & 15);
      int kb = (lane & 16) ? 8 : 0;
      v8bf lo = *(const v8bf*)(&sB[cur][n * 40 + kb]);
      v8bf hi = *(const v8bf*)(&sB[cur][n * 40 + kb + 16]);
      v16bf bfr;
#pragma unroll
      for (int j = 0; j < 8; ++j) { bfr[j] = lo[j]; bfr[j + 8] = hi[j]; }
#pragma unroll
      for (int mt = 0; mt < 4; ++mt)
        acc[mt][nt] = __builtin_amdgcn_wmma_f32_16x16x32_bf16(
            false, af[mt], false, bfr, (short)0, acc[mt][nt], false, false);
    }
    wait_async0();
    __syncthreads();
  }

  // epilogue (C layout: VGPR r -> M=r+8*(lane>=16), N=lane%16)
#pragma unroll
  for (int mt = 0; mt < 4; ++mt)
#pragma unroll
    for (int nt = 0; nt < 2; ++nt)
#pragma unroll
      for (int r = 0; r < 8; ++r) {
        int m = bm + wm * 64 + mt * 16 + r + ((lane & 16) ? 8 : 0);
        int n = bn + wn * 32 + nt * 16 + (lane & 15);
        float v = acc[mt][nt][r] + bias[n];
        size_t idx = (size_t)m * N + n;
        if (EPI == EPI_BF16) {
          ((bf16*)outp)[idx] = (bf16)v;
        } else if (EPI == EPI_BF16_GELU) {
          float gl = 0.5f * v * (1.0f + erff(v * 0.70710678118654752f));
          ((bf16*)outp)[idx] = (bf16)gl;
        } else {
          ((float*)outp)[idx] = resid[idx] + v;
        }
      }
}

// ---------------------------------------------------------------------------
// Flash attention: block = (q-tile 128, head, batch). 8 waves x 16 q-rows.
// 64-key blocks: halves barriers and online-softmax rescale rounds per key.
// ---------------------------------------------------------------------------
__global__ __launch_bounds__(256) void attn_k(const bf16* __restrict__ q,
                                              const bf16* __restrict__ k,
                                              const bf16* __restrict__ v,
                                              const float* __restrict__ bias,
                                              bf16* __restrict__ out) {
  __shared__ bf16 sK[64 * 72];        // [key][dh]
  __shared__ bf16 sVT[64 * 72];       // [dh][key]
  __shared__ bf16 sP[8][16 * 72];     // per-wave P staging (C->A relayout)

  const int tid = threadIdx.x, lane = tid & 31, wid = tid >> 5;
  const int b = blockIdx.z, h = blockIdx.y, qt = blockIdx.x;
  const int qrow0 = qt * 128 + wid * 16;
  const float scale = 0.125f;  // 1/sqrt(DH=64)
  const int mloc_off = (lane & 16) ? 8 : 0;

  v16bf aq[2];
  {
    int row = qrow0 + (lane & 15);
    int base = (lane & 16) ? 8 : 0;
    const bf16* qp = q + ((size_t)(b * Sc + row)) * Ec + h * DHc;
#pragma unroll
    for (int ks = 0; ks < 2; ++ks) {
      v8bf lo = *(const v8bf*)(qp + ks * 32 + base);
      v8bf hi = *(const v8bf*)(qp + ks * 32 + base + 16);
#pragma unroll
      for (int j = 0; j < 8; ++j) { aq[ks][j] = lo[j]; aq[ks][j + 8] = hi[j]; }
    }
  }

  float mrow[8], lrow[8];
  v8f oacc[4] = {};
#pragma unroll
  for (int r = 0; r < 8; ++r) { mrow[r] = -1e30f; lrow[r] = 0.f; }

  for (int kb = 0; kb < Sc / 64; ++kb) {
    // stage K (natural) and V (transposed) 64x64 tiles
#pragma unroll
    for (int i = 0; i < 2; ++i) {
      int c = tid + i * 256;
      int krow = c >> 3, c8 = (c & 7) << 3;
      size_t src = ((size_t)(b * Sc + kb * 64 + krow)) * Ec + h * DHc + c8;
      *(v8bf*)(&sK[krow * 72 + c8]) = *(const v8bf*)(k + src);
      v8bf vd = *(const v8bf*)(v + src);
#pragma unroll
      for (int j = 0; j < 8; ++j) sVT[(c8 + j) * 72 + krow] = vd[j];
    }
    __syncthreads();

    // S = Q @ K^T   (4 n-tiles x 2 k-steps)
    v8f sacc[4] = {};
#pragma unroll
    for (int nt = 0; nt < 4; ++nt) {
#pragma unroll
      for (int ks = 0; ks < 2; ++ks) {
        v16bf bk;
        int key = nt * 16 + (lane & 15);
        int kof = ks * 32 + ((lane & 16) ? 8 : 0);
        v8bf lo = *(const v8bf*)(&sK[key * 72 + kof]);
        v8bf hi = *(const v8bf*)(&sK[key * 72 + kof + 16]);
#pragma unroll
        for (int j = 0; j < 8; ++j) { bk[j] = lo[j]; bk[j + 8] = hi[j]; }
        sacc[nt] = __builtin_amdgcn_wmma_f32_16x16x32_bf16(
            false, aq[ks], false, bk, (short)0, sacc[nt], false, false);
      }
    }

    // online softmax + stage P
#pragma unroll
    for (int r = 0; r < 8; ++r) {
      int mloc = r + mloc_off;
      size_t brow = ((size_t)(b * Sc + qrow0 + mloc)) * Sc + kb * 64 + (lane & 15);
      float sv[4];
#pragma unroll
      for (int nt = 0; nt < 4; ++nt) sv[nt] = sacc[nt][r] * scale + bias[brow + nt * 16];
      float rm = fmaxf(fmaxf(sv[0], sv[1]), fmaxf(sv[2], sv[3]));
#pragma unroll
      for (int mk = 8; mk >= 1; mk >>= 1) rm = fmaxf(rm, __shfl_xor(rm, mk, 32));
      float mnew = fmaxf(mrow[r], rm);
      float corr = __expf(mrow[r] - mnew);
      float p[4], ps = 0.f;
#pragma unroll
      for (int nt = 0; nt < 4; ++nt) { p[nt] = __expf(sv[nt] - mnew); ps += p[nt]; }
#pragma unroll
      for (int mk = 8; mk >= 1; mk >>= 1) ps += __shfl_xor(ps, mk, 32);
      lrow[r] = lrow[r] * corr + ps;
      mrow[r] = mnew;
#pragma unroll
      for (int nt = 0; nt < 4; ++nt) oacc[nt][r] *= corr;
#pragma unroll
      for (int nt = 0; nt < 4; ++nt)
        sP[wid][mloc * 72 + nt * 16 + (lane & 15)] = (bf16)p[nt];
    }

    // O += P @ V   (2 k-steps over 64 keys, 4 n-tiles over dh)
#pragma unroll
    for (int ks = 0; ks < 2; ++ks) {
      v16bf ap;
      int row = lane & 15;
      int base = ks * 32 + ((lane & 16) ? 8 : 0);
      v8bf lo = *(const v8bf*)(&sP[wid][row * 72 + base]);
      v8bf hi = *(const v8bf*)(&sP[wid][row * 72 + base + 16]);
#pragma unroll
      for (int j = 0; j < 8; ++j) { ap[j] = lo[j]; ap[j + 8] = hi[j]; }
#pragma unroll
      for (int nt = 0; nt < 4; ++nt) {
        v16bf bv;
        int dh = nt * 16 + (lane & 15);
        int kof = ks * 32 + ((lane & 16) ? 8 : 0);
        v8bf blo = *(const v8bf*)(&sVT[dh * 72 + kof]);
        v8bf bhi = *(const v8bf*)(&sVT[dh * 72 + kof + 16]);
#pragma unroll
        for (int j = 0; j < 8; ++j) { bv[j] = blo[j]; bv[j + 8] = bhi[j]; }
        oacc[nt] = __builtin_amdgcn_wmma_f32_16x16x32_bf16(
            false, ap, false, bv, (short)0, oacc[nt], false, false);
      }
    }
    __syncthreads();
  }

#pragma unroll
  for (int nt = 0; nt < 4; ++nt)
#pragma unroll
    for (int r = 0; r < 8; ++r) {
      int mloc = r + mloc_off;
      float val = oacc[nt][r] / lrow[r];
      out[((size_t)(b * Sc + qrow0 + mloc)) * Ec + h * DHc + nt * 16 + (lane & 15)] =
          (bf16)val;
    }
}

// ---------------------------------------------------------------------------
extern "C" void kernel_launch(void* const* d_in, const int* in_sizes, int n_in,
                              void* d_out, int out_size, void* d_ws, size_t ws_size,
                              hipStream_t stream) {
  (void)in_sizes; (void)n_in; (void)out_size; (void)ws_size;
  const float* src = (const float*)d_in[0];
  const float* sb  = (const float*)d_in[1];
  const float* db  = (const float*)d_in[2];
  const float* Wq  = (const float*)d_in[3];
  const float* bq  = (const float*)d_in[4];
  const float* Wk  = (const float*)d_in[5];
  const float* bk  = (const float*)d_in[6];
  const float* Wv  = (const float*)d_in[7];
  const float* bv  = (const float*)d_in[8];
  const float* Wo  = (const float*)d_in[9];
  const float* bo  = (const float*)d_in[10];
  const float* g1  = (const float*)d_in[11];
  const float* be1 = (const float*)d_in[12];
  const float* g2  = (const float*)d_in[13];
  const float* be2 = (const float*)d_in[14];
  const float* W1  = (const float*)d_in[15];
  const float* bf1 = (const float*)d_in[16];
  const float* W2  = (const float*)d_in[17];
  const float* bf2 = (const float*)d_in[18];
  const float* gf  = (const float*)d_in[19];
  const float* bef = (const float*)d_in[20];

  char* ws = (char*)d_ws;
  const size_t MBy = 1024u * 1024u;
  float* x      = (float*)(ws + 0);          // 16 MB running activation
  bf16*  hbuf   = (bf16*)(ws + 16 * MBy);    // 8 MB LN output
  bf16*  qb     = (bf16*)(ws + 24 * MBy);    // 8 MB
  bf16*  kbuf   = (bf16*)(ws + 32 * MBy);    // 8 MB
  bf16*  vbuf   = (bf16*)(ws + 40 * MBy);    // 8 MB
  bf16*  attn   = (bf16*)(ws + 48 * MBy);    // 8 MB
  bf16*  ffn    = (bf16*)(ws + 24 * MBy);    // 32 MB, reuses q/k/v/attn region
  float* biasum = (float*)(ws + 56 * MBy);   // 16 MB
  bf16*  WqT    = (bf16*)(ws + 72 * MBy);    // 2 MB each, per-layer transposed
  bf16*  WkT    = (bf16*)(ws + 74 * MBy);
  bf16*  WvT    = (bf16*)(ws + 76 * MBy);
  bf16*  WoT    = (bf16*)(ws + 78 * MBy);
  bf16*  W1T    = (bf16*)(ws + 80 * MBy);    // 8 MB  [F][E]
  bf16*  W2T    = (bf16*)(ws + 88 * MBy);    // 8 MB  [E][F]

  const int M = Bc * Sc;

  hipMemcpyAsync(x, src, (size_t)M * Ec * sizeof(float),
                 hipMemcpyDeviceToDevice, stream);
  {
    int n = Bc * Sc * Sc;
    add_bias_k<<<(n + 255) / 256, 256, 0, stream>>>(sb, db, biasum, n);
  }

  dim3 blk(256);
  dim3 gEE(Ec / 128, M / 128);
  dim3 gEF(Fc / 128, M / 128);
  dim3 gAttn(Sc / 128, Hc, Bc);
  dim3 gTee(Ec / 64, Ec / 64);
  dim3 gTef(Ec / 64, Fc / 64);   // W1: K=E, N=F
  dim3 gTfe(Fc / 64, Ec / 64);   // W2: K=F, N=E

  for (int i = 0; i < Lc; ++i) {
    transpose_cvt_k<<<gTee, blk, 0, stream>>>(Wq + (size_t)i * Ec * Ec, WqT, Ec, Ec);
    transpose_cvt_k<<<gTee, blk, 0, stream>>>(Wk + (size_t)i * Ec * Ec, WkT, Ec, Ec);
    transpose_cvt_k<<<gTee, blk, 0, stream>>>(Wv + (size_t)i * Ec * Ec, WvT, Ec, Ec);
    transpose_cvt_k<<<gTee, blk, 0, stream>>>(Wo + (size_t)i * Ec * Ec, WoT, Ec, Ec);
    transpose_cvt_k<<<gTef, blk, 0, stream>>>(W1 + (size_t)i * Ec * Fc, W1T, Ec, Fc);
    transpose_cvt_k<<<gTfe, blk, 0, stream>>>(W2 + (size_t)i * Fc * Ec, W2T, Fc, Ec);

    layernorm_k<1><<<M, blk, 0, stream>>>(x, g1 + i * Ec, be1 + i * Ec, hbuf);
    gemm_bf16<EPI_BF16><<<gEE, blk, 0, stream>>>(hbuf, WqT, bq + i * Ec, nullptr, qb, M, Ec, Ec);
    gemm_bf16<EPI_BF16><<<gEE, blk, 0, stream>>>(hbuf, WkT, bk + i * Ec, nullptr, kbuf, M, Ec, Ec);
    gemm_bf16<EPI_BF16><<<gEE, blk, 0, stream>>>(hbuf, WvT, bv + i * Ec, nullptr, vbuf, M, Ec, Ec);
    attn_k<<<gAttn, blk, 0, stream>>>(qb, kbuf, vbuf, biasum, attn);
    gemm_bf16<EPI_F32_RESID><<<gEE, blk, 0, stream>>>(attn, WoT, bo + i * Ec, x, x, M, Ec, Ec);
    layernorm_k<1><<<M, blk, 0, stream>>>(x, g2 + i * Ec, be2 + i * Ec, hbuf);
    gemm_bf16<EPI_BF16_GELU><<<gEF, blk, 0, stream>>>(hbuf, W1T, bf1 + i * Fc, nullptr, ffn, M, Fc, Ec);
    gemm_bf16<EPI_F32_RESID><<<gEE, blk, 0, stream>>>(ffn, W2T, bf2 + i * Ec, x, x, M, Ec, Fc);
  }
  layernorm_k<0><<<M, blk, 0, stream>>>(x, gf, bef, d_out);
}
